// AdditiveAttention_1803886265686
// MI455X (gfx1250) — matrix operand order repaired and checked by reading, not compiled, not run on hardware
//
#include <hip/hip_runtime.h>

typedef __attribute__((ext_vector_type(16))) _Float16 v16h;
typedef __attribute__((ext_vector_type(8)))  float    v8f;

#define B_  8
#define Q_  1024
#define K_  1024
#define H_  64
#define NEG_MASK (-1000000.0f)

#define QSTRIDE 65   // q tile pad: banks (m+h)%64 -> conflict-free score reads
#define VSTRIDE 68   // k/v tile pad: 272B rows, 16B-aligned for async b128 staging
#define NWAVE   4    // waves per attention block (4 q-tiles share k/v staging)
#define NT      (K_ / 32)

#if __has_builtin(__builtin_amdgcn_global_load_async_to_lds_b128)
#define HAVE_ASYNC_LDS 1
typedef int v4i __attribute__((ext_vector_type(4)));
typedef __attribute__((address_space(1))) v4i* gptr_v4i;   // global int4*
typedef __attribute__((address_space(3))) v4i* lptr_v4i;   // LDS int4*
#else
#define HAVE_ASYNC_LDS 0
#endif

__device__ __forceinline__ float fast_tanh(float x) {
#if __has_builtin(__builtin_amdgcn_tanhf)
  return __builtin_amdgcn_tanhf(x);          // v_tanh_f32 (gfx1250 TRANS op)
#else
  float e = __expf(-2.0f * __builtin_fabsf(x));
  float t = __fdividef(1.0f - e, 1.0f + e);
  return __builtin_copysignf(t, x);
#endif
}

// ---------------------------------------------------------------------------
// Kernel 1: fused projection  qp = queries @ W_q,  kp = keys @ W_k
// One wave per 16x16 output tile, K-loop of two v_wmma_f32_16x16x32_f16.
// ---------------------------------------------------------------------------
__global__ __launch_bounds__(128) void proj_kernel(
    const float* __restrict__ queries, const float* __restrict__ keys,
    const float* __restrict__ Wq,      const float* __restrict__ Wk,
    float* __restrict__ qp,            float* __restrict__ kp) {
  const int wave = threadIdx.x >> 5;
  const int lane = threadIdx.x & 31;
  const int t    = blockIdx.x * 4 + wave;          // 0..4095

  const float* X; const float* W; float* Y;
  if (t < 2048) { X = queries; W = Wq; Y = qp; }
  else          { X = keys;    W = Wk; Y = kp; }
  const int tl = t & 2047;
  const int rt = tl >> 2;                          // 512 row tiles of 8192 rows
  const int ct = tl & 3;                           // 4 col tiles of 64 cols
  const int m  = lane & 15;
  const int hi = lane >> 4;

  v8f acc = {};
  for (int kk = 0; kk < H_; kk += 32) {
    v16h a, b;
    const float* xrow = X + (size_t)(rt * 16 + m) * H_ + kk;
    #pragma unroll
    for (int i = 0; i < 16; ++i) {                 // A layout: K = (i<8?i:i+8)+8*hi
      int K = (i < 8 ? i : i + 8) + 8 * hi;
      a[i] = (_Float16)xrow[K];
    }
    #pragma unroll
    for (int i = 0; i < 16; ++i) {                 // B layout: K = (i<8?i:i+8)+8*hi, N = lane%16
      int K = (i < 8 ? i : i + 8) + 8 * hi;
      b[i] = (_Float16)W[(size_t)(kk + K) * H_ + ct * 16 + m];
    }
    acc = __builtin_amdgcn_wmma_f32_16x16x32_f16(false, a, false, b,
                                                 (short)0, acc, false, false);
  }
  float* yrow = Y + (size_t)(rt * 16) * H_ + ct * 16;
  #pragma unroll
  for (int r = 0; r < 8; ++r)                      // C layout: M = r+8*hi, N = lane%16
    yrow[(size_t)(r + 8 * hi) * H_ + m] = acc[r];
}

// ---------------------------------------------------------------------------
// Stage a 32x64 fp32 tile pair (kp, V) into padded LDS.
// Async path: GLOBAL_LOAD_ASYNC_TO_LDS_B128, 4+4 chunks per thread (128 thr).
// ---------------------------------------------------------------------------
__device__ __forceinline__ void stage_tiles(const float* __restrict__ ksrc,
                                            const float* __restrict__ vsrc,
                                            float* kdst, float* vdst, int tid) {
#if HAVE_ASYNC_LDS
  #pragma unroll
  for (int c = 0; c < 4; ++c) {
    int idx = tid + c * 128;                       // 0..511 16B-chunks
    int r = idx >> 4, ch = idx & 15;
    __builtin_amdgcn_global_load_async_to_lds_b128(
        (gptr_v4i)(float*)(ksrc + (size_t)r * H_ + ch * 4),
        (lptr_v4i)(kdst + r * VSTRIDE + ch * 4), 0, 0);
    __builtin_amdgcn_global_load_async_to_lds_b128(
        (gptr_v4i)(float*)(vsrc + (size_t)r * H_ + ch * 4),
        (lptr_v4i)(vdst + r * VSTRIDE + ch * 4), 0, 0);
  }
#else
  for (int i = tid; i < 32 * H_; i += 128) {
    int r = i >> 6, c = i & 63;
    kdst[r * VSTRIDE + c] = ksrc[(size_t)r * H_ + c];
    vdst[r * VSTRIDE + c] = vsrc[(size_t)r * H_ + c];
  }
#endif
}

__device__ __forceinline__ void wait_stage() {
#if HAVE_ASYNC_LDS
  asm volatile("s_wait_asynccnt 0x0" ::: "memory");
#endif
}

// ---------------------------------------------------------------------------
// Kernel 2: fused scores(tanh) + masked online-softmax + P@V.
// 4 waves/block = 4 q-tiles of one batch sharing double-buffered k/v tiles.
// ---------------------------------------------------------------------------
__global__ __launch_bounds__(128) void attn_kernel(
    const float* __restrict__ qp, const float* __restrict__ kp,
    const float* __restrict__ values, const int* __restrict__ valid_lens,
    const float* __restrict__ w_v, float* __restrict__ out) {
  __shared__ float qs[NWAVE][16 * QSTRIDE];
  __shared__ float kbuf[2][32 * VSTRIDE];
  __shared__ float vbuf[2][32 * VSTRIDE];
  __shared__ float wS[H_];
  __shared__ float scaleS[NWAVE][16];
  __shared__ float lS[NWAVE][16];

  const int tid  = threadIdx.x;
  const int wave = tid >> 5;
  const int lane = tid & 31;
  const int b    = blockIdx.x >> 4;                // 16 blocks per batch
  const int qt   = (blockIdx.x & 15) * NWAVE + wave;
  const int m    = lane & 15;
  const int hi   = lane >> 4;
  const int vlen = valid_lens[b];

  // stage w_v and this wave's q-projection tile
  if (tid < H_) wS[tid] = w_v[tid];
  const float* qsrc = qp + ((size_t)b * Q_ + (size_t)qt * 16) * H_;
  for (int i = lane; i < 16 * H_; i += 32) {
    int r = i >> 6, c = i & 63;
    qs[wave][r * QSTRIDE + c] = qsrc[(size_t)r * H_ + c];
  }

  // per-lane WMMA-A column mapping for this lane's 16 score slots
  int kcols[16];
  #pragma unroll
  for (int i = 0; i < 16; ++i)
    kcols[i] = (i < 8 ? i : i + 8) + 8 * hi;

  float mrow = -1e30f, lrow = 0.0f;
  v8f acc[4] = {{}, {}, {}, {}};                   // 16x64 accumulator (C layout)

  const float* kbase = kp     + (size_t)b * K_ * H_;
  const float* vbase = values + (size_t)b * K_ * H_;

  stage_tiles(kbase, vbase, kbuf[0], vbuf[0], tid);   // prologue: tile 0

  for (int kt = 0; kt < NT; ++kt) {
    const int cur = kt & 1;
    wait_stage();                                  // current buffer landed
    __syncthreads();                               // visible to all waves;
                                                   // other buffer free for reuse
    if (kt + 1 < NT)                               // prefetch next tile async
      stage_tiles(kbase + (size_t)(kt + 1) * 32 * H_,
                  vbase + (size_t)(kt + 1) * 32 * H_,
                  kbuf[1 - cur], vbuf[1 - cur], tid);

    const float* ksb = kbuf[cur];
    const float* vsb = vbuf[cur];

    // scores: s[i] = sum_h w[h] * tanh(qp[m,h] + kp[kcols[i],h])  -- hot loop
    float s[16];
    #pragma unroll
    for (int i = 0; i < 16; ++i) s[i] = 0.0f;
    for (int h = 0; h < H_; ++h) {
      float qv = qs[wave][m * QSTRIDE + h];        // conflict-free (pad 65)
      float wv = wS[h];                            // uniform broadcast
      #pragma unroll
      for (int i = 0; i < 16; ++i) {               // 2 distinct rows -> broadcast
        float tv = fast_tanh(qv + ksb[kcols[i] * VSTRIDE + h]);
        s[i] = __builtin_fmaf(wv, tv, s[i]);
      }
    }

    // mask + tile max
    const int kg0 = kt * 32;
    float tmax = -1e30f;
    #pragma unroll
    for (int i = 0; i < 16; ++i) {
      if (kg0 + kcols[i] >= vlen) s[i] = NEG_MASK;
      tmax = fmaxf(tmax, s[i]);
    }
    tmax = fmaxf(tmax, __shfl_xor(tmax, 16));      // join the two half-rows

    // online softmax update
    float mnew  = fmaxf(mrow, tmax);
    float scale = __expf(mrow - mnew);
    mrow = mnew;
    float psum = 0.0f;
    v16h pa;                                       // probs in WMMA-A f16 layout
    #pragma unroll
    for (int i = 0; i < 16; ++i) {
      float p = __expf(s[i] - mnew);
      psum += p;
      pa[i] = (_Float16)p;
    }
    psum += __shfl_xor(psum, 16);
    lrow = lrow * scale + psum;

    // broadcast row rescale factors to the C-layout lanes
    if (lane < 16) scaleS[wave][lane] = scale;
    __syncthreads();                               // uniform across all 4 waves
    float sc[8];
    #pragma unroll
    for (int r = 0; r < 8; ++r) sc[r] = scaleS[wave][r + 8 * hi];
    #pragma unroll
    for (int nt = 0; nt < 4; ++nt)
      #pragma unroll
      for (int r = 0; r < 8; ++r) acc[nt][r] *= sc[r];

    // P(16x32) @ V(32x64) via four v_wmma_f32_16x16x32_f16
    #pragma unroll
    for (int nt = 0; nt < 4; ++nt) {
      v16h bv;
      #pragma unroll
      for (int i = 0; i < 16; ++i) {               // B layout: K = (i<8?i:i+8)+8*hi, N = m
        int K = (i < 8 ? i : i + 8) + 8 * hi;
        bv[i] = (_Float16)vsb[K * VSTRIDE + nt * 16 + m];
      }
      acc[nt] = __builtin_amdgcn_wmma_f32_16x16x32_f16(false, pa, false, bv,
                                                       (short)0, acc[nt], false, false);
    }
  }

  // normalize and store
  if (lane < 16) lS[wave][lane] = lrow;
  __syncthreads();
  float* obase = out + ((size_t)b * Q_ + (size_t)qt * 16) * H_;
  #pragma unroll
  for (int r = 0; r < 8; ++r) {
    int row = r + 8 * hi;
    float inv = __fdividef(1.0f, lS[wave][row]);
    #pragma unroll
    for (int nt = 0; nt < 4; ++nt)
      obase[(size_t)row * H_ + nt * 16 + m] = acc[nt][r] * inv;
  }
}

// ---------------------------------------------------------------------------
extern "C" void kernel_launch(void* const* d_in, const int* in_sizes, int n_in,
                              void* d_out, int out_size, void* d_ws, size_t ws_size,
                              hipStream_t stream) {
  const float* queries = (const float*)d_in[0];
  const float* keys    = (const float*)d_in[1];
  const float* values  = (const float*)d_in[2];
  const int*   vlens   = (const int*)  d_in[3];
  const float* Wq      = (const float*)d_in[4];
  const float* Wk      = (const float*)d_in[5];
  const float* wv      = (const float*)d_in[6];
  float* out = (float*)d_out;

  float* qp = (float*)d_ws;                              // [8192, 64] fp32 (2 MB)
  float* kp = qp + (size_t)B_ * K_ * H_;                 // [8192, 64] fp32 (2 MB)

  // 4096 16x16 tiles, 4 waves/block -> 1024 blocks
  proj_kernel<<<1024, 128, 0, stream>>>(queries, keys, Wq, Wk, qp, kp);
  // 4 q-tiles per block sharing k/v staging: 8 batches * 16 blocks = 128 blocks
  attn_kernel<<<128, 128, 0, stream>>>(qp, kp, values, vlens, wv, out);
}
